// BEVNet_80934363726456
// MI455X (gfx1250) — matrix operand order repaired and checked by reading, not compiled, not run on hardware
//
#include <hip/hip_runtime.h>

// ---------------------------------------------------------------------------
// Types for CDNA5 WMMA (wave32)
// ---------------------------------------------------------------------------
typedef __attribute__((ext_vector_type(16))) _Float16 v16h;
typedef __attribute__((ext_vector_type(8)))  _Float16 v8h;
typedef __attribute__((ext_vector_type(8)))  float    v8f;

#define NT 4  // 16-wide N tiles per wave
#define MT 2  // 16-tall M tiles per wave (32 couts) -- halves B traffic

// Fragment cin-lane mapping (ISA 7.12.2, f16):
//  A 16x32: lane(half,idx): m=idx, element e -> k = (e<8 ? e : e+8) + 8*half
//           => contiguous cin runs [8h..8h+7] and [16+8h..16+8h+7]
//  B 32x16: lane(half,idx): n=idx, element e -> k = (e<8 ? e : e+8) + 8*half
//           => contiguous cin runs [8h..8h+7] and [16+8h..16+8h+7]
// Both runs are contiguous in LDS when staged cin-fastest -> ds_load_b128 pairs.

// ---------------------------------------------------------------------------
// Implicit-GEMM conv (NCHW, OIHW weights), tap-outer K ordering, LDS staging.
// GEMM: M=Cout, N=Hout*Wout, K=Cin*KH*KW.  One wave -> 32(M) x 64(N) tile.
// Requirements (all layers satisfy): Cout%32==0, HW%64==0, Wout%64==0, Cin%32==0.
// ---------------------------------------------------------------------------
template <int KH, int KW, int STRIDE>
__global__ __launch_bounds__(128, 2)
void conv_wmma_t(const float* __restrict__ x, const float* __restrict__ w,
                 const float* res, float* y,
                 int Cin, int Hin, int Win, int Cout, int Hout, int Wout,
                 int pad, int doRelu)
{
  constexpr int NCOLS = 63 * STRIDE + KW;          // input cols needed for 64 outputs
  __shared__ _Float16 sB[4][32 * NCOLS];           // [wave][col*32 + cin]
  __shared__ _Float16 sA[4][KW * 32 * 32];         // [wave][(kx*32+m)*32 + cin]

  const int lane = threadIdx.x & 31;
  const int wv   = threadIdx.x >> 5;
  const int wave = blockIdx.x * 4 + wv;
  const int HW = Hout * Wout;
  const int mSuper  = Cout >> 5;                   // 32-row M supertiles
  const int nGroups = HW >> 6;
  if (wave >= mSuper * nGroups) return;            // wave-uniform exit
  const int mi = wave % mSuper;
  const int ng = wave / mSuper;
  const int m0 = mi << 5;
  const int n0 = ng << 6;
  const int half = lane >> 4;
  const int idx  = lane & 15;
  const int KK = KH * KW;
  const int K  = Cin * KK;
  const int oy  = n0 / Wout;                       // 64-chunk never crosses a row
  const int ox0 = n0 - oy * Wout;
  const int HWin = Hin * Win;

  v8f acc[MT][NT] = {};

  for (int ky = 0; ky < KH; ++ky) {
    const int iy = oy * STRIDE + ky - pad;
    const bool rowOk = (unsigned)iy < (unsigned)Hin;
    const int colBase = ox0 * STRIDE - pad;
    const float* xrow = x + (size_t)iy * Win;

    for (int cin0 = 0; cin0 < Cin; cin0 += 32) {
      // ---- stage input strip: 32 cin x NCOLS cols, f32 -> f16, cin-fastest ----
      for (int id = lane; id < 32 * NCOLS; id += 32) {
        int j  = id % NCOLS;                       // consecutive lanes -> consecutive cols
        int ci = id / NCOLS;
        int ix = colBase + j;
        float v = 0.0f;
        if (rowOk && (unsigned)ix < (unsigned)Win)
          v = xrow[(size_t)(cin0 + ci) * HWin + ix];
        sB[wv][j * 32 + ci] = (_Float16)v;
      }
      // ---- stage weight slab: KW x 32 m x 32 cin ----
      for (int id = lane; id < KW * 32 * 32; id += 32) {
        int ci = id & 31;
        int mm = (id >> 5) & 31;
        int kx = id >> 10;
        float v = w[(size_t)(m0 + mm) * K + (size_t)(cin0 + ci) * KK + ky * KW + kx];
        sA[wv][(kx * 32 + mm) * 32 + ci] = (_Float16)v;
      }
      // ---- compute: KW taps x (MT x NT) WMMAs, fragments via ds_load_b128 ----
#pragma unroll
      for (int kx = 0; kx < KW; ++kx) {
        v16h a[MT];
#pragma unroll
        for (int u = 0; u < MT; ++u) {
          const _Float16* aBase = &sA[wv][(kx * 32 + u * 16 + idx) * 32];
          v8h alo = *(const v8h*)(aBase + 8 * half);
          v8h ahi = *(const v8h*)(aBase + 16 + 8 * half);
          a[u] = __builtin_shufflevector(alo, ahi,
                 0,1,2,3,4,5,6,7,8,9,10,11,12,13,14,15);
        }
#pragma unroll
        for (int t = 0; t < NT; ++t) {
          int col = (t * 16 + idx) * STRIDE + kx;
          const _Float16* bBase = &sB[wv][col * 32];
          v8h blo = *(const v8h*)(bBase + 8 * half);
          v8h bhi = *(const v8h*)(bBase + 16 + 8 * half);
          v16h b = __builtin_shufflevector(blo, bhi,
                   0,1,2,3,4,5,6,7,8,9,10,11,12,13,14,15);
#pragma unroll
          for (int u = 0; u < MT; ++u) {
            acc[u][t] = __builtin_amdgcn_wmma_f32_16x16x32_f16(
                false, a[u], false, b, (short)0, acc[u][t], false, false);
          }
        }
      }
    }
  }

  // D layout: VGPR r -> M = m0 + u*16 + r + 8*half, N = n0 + t*16 + (lane&15)
#pragma unroll
  for (int u = 0; u < MT; ++u) {
#pragma unroll
    for (int t = 0; t < NT; ++t) {
      int n = n0 + t * 16 + idx;
#pragma unroll
      for (int r = 0; r < 8; ++r) {
        int m = m0 + u * 16 + r + half * 8;
        size_t off = (size_t)m * HW + n;
        float v = acc[u][t][r];
        if (res) v += res[off];
        if (doRelu) v = fmaxf(v, 0.0f);
        y[off] = v;
      }
    }
  }
}

// ---------------------------------------------------------------------------
// Monotone float<->uint flip so unsigned atomicMax == float max
// ---------------------------------------------------------------------------
__device__ __forceinline__ unsigned flipf(float f) {
  unsigned u = __float_as_uint(f);
  return (u & 0x80000000u) ? ~u : (u | 0x80000000u);
}
__device__ __forceinline__ float unflipf(unsigned v) {
  unsigned u = (v & 0x80000000u) ? (v & 0x7FFFFFFFu) : ~v;
  return __uint_as_float(u);
}

__global__ void vox_init_kernel(unsigned* g, int n) {
  int i = blockIdx.x * blockDim.x + threadIdx.x;
  if (i < n) g[i] = 0x007FFFFFu;  // flip(-inf)
}

__global__ void vox_finalize_kernel(const unsigned* __restrict__ vox,
                                    float* __restrict__ out, int n) {
  int i = blockIdx.x * blockDim.x + threadIdx.x;
  if (i >= n) return;
  float v = unflipf(vox[i]);
  unsigned e = __float_as_uint(v) & 0x7F800000u;
  out[i] = (e == 0x7F800000u) ? 0.0f : v;   // -inf (empty) / nan -> 0
}

// Fused bilinear_sample(feat, scoord, sscale) -> voxel max scatter
__global__ void sample_voxmax_kernel(const float* __restrict__ feat,
                                     int C, int H, int W,
                                     const float* __restrict__ scoord, float sscale,
                                     const float* __restrict__ pind, float pscale,
                                     unsigned* __restrict__ vox, int Ho, int Wo, int N)
{
  int n = blockIdx.x * blockDim.x + threadIdx.x;
  if (n >= N) return;
  float h  = fminf(fmaxf(scoord[2 * n + 0] * sscale, 0.0f), (float)(H - 1));
  float wv = fminf(fmaxf(scoord[2 * n + 1] * sscale, 0.0f), (float)(W - 1));
  int h0 = min(max((int)floorf(h), 0), H - 2);
  int w0 = min(max((int)floorf(wv), 0), W - 2);
  float dh = h - (float)h0, dw = wv - (float)w0;
  float w00 = (1.0f - dh) * (1.0f - dw), w01 = (1.0f - dh) * dw;
  float w10 = dh * (1.0f - dw),          w11 = dh * dw;
  int i00 = h0 * W + w0;

  int hi = min(max((int)floorf(pind[2 * n + 0] * pscale), 0), Ho - 1);
  int wi = min(max((int)floorf(pind[2 * n + 1] * pscale), 0), Wo - 1);
  int flat = hi * Wo + wi;
  int HWf = H * W, HWo = Ho * Wo;
  for (int c = 0; c < C; ++c) {
    const float* f = feat + (size_t)c * HWf;
    float v = f[i00] * w00 + f[i00 + 1] * w01 + f[i00 + W] * w10 + f[i00 + W + 1] * w11;
    atomicMax(&vox[(size_t)c * HWo + flat], flipf(v));
  }
}

// Plain bilinear point sampling -> out[C][N]
__global__ void sample_points_kernel(const float* __restrict__ feat,
                                     int C, int H, int W,
                                     const float* __restrict__ coord, float scale,
                                     float* __restrict__ out, int N)
{
  int n = blockIdx.x * blockDim.x + threadIdx.x;
  if (n >= N) return;
  float h  = fminf(fmaxf(coord[2 * n + 0] * scale, 0.0f), (float)(H - 1));
  float wv = fminf(fmaxf(coord[2 * n + 1] * scale, 0.0f), (float)(W - 1));
  int h0 = min(max((int)floorf(h), 0), H - 2);
  int w0 = min(max((int)floorf(wv), 0), W - 2);
  float dh = h - (float)h0, dw = wv - (float)w0;
  float w00 = (1.0f - dh) * (1.0f - dw), w01 = (1.0f - dh) * dw;
  float w10 = dh * (1.0f - dw),          w11 = dh * dw;
  int i00 = h0 * W + w0;
  int HWf = H * W;
  for (int c = 0; c < C; ++c) {
    const float* f = feat + (size_t)c * HWf;
    out[(size_t)c * N + n] =
        f[i00] * w00 + f[i00 + 1] * w01 + f[i00 + W] * w10 + f[i00 + W + 1] * w11;
  }
}

// Bilinear resize (align-corners style, matches resize_ac)
__global__ void resize_kernel(const float* __restrict__ in, int C, int H, int W,
                              float* __restrict__ out, int Ho, int Wo)
{
  int id = blockIdx.x * blockDim.x + threadIdx.x;
  int total = C * Ho * Wo;
  if (id >= total) return;
  int ox = id % Wo;
  int t  = id / Wo;
  int oy = t % Ho;
  int c  = t / Ho;
  float ys = (float)oy * (float)(H - 1) / (float)(Ho - 1);
  float xs = (float)ox * (float)(W - 1) / (float)(Wo - 1);
  int y0 = min(max((int)floorf(ys), 0), H - 2);
  int x0 = min(max((int)floorf(xs), 0), W - 2);
  float wy = ys - (float)y0, wx = xs - (float)x0;
  const float* f = in + (size_t)c * H * W;
  float a = f[y0 * W + x0]       * (1.0f - wy) + f[(y0 + 1) * W + x0]     * wy;
  float b = f[y0 * W + x0 + 1]   * (1.0f - wy) + f[(y0 + 1) * W + x0 + 1] * wy;
  out[id] = a * (1.0f - wx) + b * wx;
}

// 1x1 conv, Cout=3, + bias (aux heads)
__global__ void aux_head_kernel(const float* __restrict__ x, int Cin, int HW,
                                const float* __restrict__ w, const float* __restrict__ b,
                                float* __restrict__ out)
{
  int p = blockIdx.x * blockDim.x + threadIdx.x;
  if (p >= HW) return;
  float a0 = b[0], a1 = b[1], a2 = b[2];
  for (int ci = 0; ci < Cin; ++ci) {
    float xv = x[(size_t)ci * HW + p];
    a0 += w[ci] * xv;
    a1 += w[Cin + ci] * xv;
    a2 += w[2 * Cin + ci] * xv;
  }
  out[p] = a0;
  out[(size_t)HW + p] = a1;
  out[(size_t)2 * HW + p] = a2;
}

__global__ void add_inplace_kernel(float* __restrict__ a,
                                   const float* __restrict__ b, int n) {
  int i = blockIdx.x * blockDim.x + threadIdx.x;
  if (i < n) a[i] += b[i];
}

// ---------------------------------------------------------------------------
// Host side
// ---------------------------------------------------------------------------
static void launch_conv(const float* x, const float* w, const float* res, float* y,
                        int Cin, int Hin, int Win, int Cout, int Hout, int Wout,
                        int KH, int KW, int stride, int pad, int relu, hipStream_t s)
{
  int HW = Hout * Wout;
  int mSuper = Cout / 32;
  int nGroups = HW / 64;
  int tiles = mSuper * nGroups;
  int blocks = (tiles + 3) / 4;
  if (KH == 3 && stride == 2) {
    conv_wmma_t<3, 3, 2><<<blocks, 128, 0, s>>>(x, w, res, y, Cin, Hin, Win,
                                                Cout, Hout, Wout, pad, relu);
  } else if (KH == 3) {
    conv_wmma_t<3, 3, 1><<<blocks, 128, 0, s>>>(x, w, res, y, Cin, Hin, Win,
                                                Cout, Hout, Wout, pad, relu);
  } else {
    conv_wmma_t<1, 1, 1><<<blocks, 128, 0, s>>>(x, w, res, y, Cin, Hin, Win,
                                                Cout, Hout, Wout, pad, relu);
  }
}

extern "C" void kernel_launch(void* const* d_in, const int* in_sizes, int n_in,
                              void* d_out, int out_size, void* d_ws, size_t ws_size,
                              hipStream_t stream)
{
  (void)n_in; (void)out_size; (void)ws_size;
  // ---- inputs (setup_inputs dict order, params flattened in literal order) ----
  const float* c      = (const float*)d_in[0];   // (192,512,512)
  const float* ccoord = (const float*)d_in[1];   // (N,2)
  const float* pcoord = (const float*)d_in[2];   // (N,2)
  const float* deep   = (const float*)d_in[3];   // (128,64,64)
  const int N = in_sizes[1] / 2;

  int pi = 4;
  const float* ch_down = (const float*)d_in[pi++];
  const float* ch_b[6];  for (int i = 0; i < 6;  ++i) ch_b[i]  = (const float*)d_in[pi++];
  const float* ph_down = (const float*)d_in[pi++];
  const float* ph_b[4];  for (int i = 0; i < 4;  ++i) ph_b[i]  = (const float*)d_in[pi++];
  const float* cr1_down = (const float*)d_in[pi++];
  const float* cr1_b[8]; for (int i = 0; i < 8;  ++i) cr1_b[i] = (const float*)d_in[pi++];
  const float* pr1_down = (const float*)d_in[pi++];
  const float* pr1_b[6]; for (int i = 0; i < 6;  ++i) pr1_b[i] = (const float*)d_in[pi++];
  const float* cr2_down = (const float*)d_in[pi++];
  const float* cr2_b[10];for (int i = 0; i < 10; ++i) cr2_b[i] = (const float*)d_in[pi++];
  const float* w_fuse = (const float*)d_in[pi++];
  const float* w_oc1  = (const float*)d_in[pi++];
  const float* w_oc2  = (const float*)d_in[pi++];
  const float* aux1w = (const float*)d_in[pi++]; const float* aux1b = (const float*)d_in[pi++];
  const float* aux2w = (const float*)d_in[pi++]; const float* aux2b = (const float*)d_in[pi++];
  const float* aux3w = (const float*)d_in[pi++]; const float* aux3b = (const float*)d_in[pi++];

  // ---- workspace layout (floats) ----
  float* W = (float*)d_ws;
  size_t o = 0;
  auto alloc = [&](size_t nf) { float* p = W + o; o += nf; return p; };
  const int HW256 = 256 * 256, HW128 = 128 * 128, HW64 = 64 * 64;
  const int HWp0 = 32 * 1024, HWp1 = 16 * 512;

  float* c0cat  = alloc((size_t)64 * HW256);    // [c0 | p0_to_cart]
  float* tmpBig = alloc((size_t)128 * HW256);   // shared conv temp / out_conv1 result
  float* c0p    = alloc((size_t)32 * HWp0);     // c0_to_polar
  float* p0     = alloc((size_t)32 * HWp0);
  float* c1cat  = alloc((size_t)128 * HW128);   // [c1 | p1_to_cart]
  float* c1p    = alloc((size_t)64 * HWp1);
  float* p1     = alloc((size_t)64 * HWp1);
  float* c2a    = alloc((size_t)128 * HW64);
  float* c2b    = alloc((size_t)128 * HW64);
  float* cat320 = alloc((size_t)320 * HW256);
  unsigned* vox = (unsigned*)alloc((size_t)32 * HW256);  // max voxel grid: 32x256x256

  const int TPB = 256;
  auto blk = [&](int n) { return (n + TPB - 1) / TPB; };

  // ---- cart_header: 192->32, stride 2, 512^2 -> 256^2, + 3 blocks ----
  launch_conv(c, ch_down, nullptr, c0cat, 192, 512, 512, 32, 256, 256, 3, 3, 2, 1, 1, stream);
  for (int i = 0; i < 3; ++i) {
    launch_conv(c0cat, ch_b[2 * i],     nullptr, tmpBig, 32, 256, 256, 32, 256, 256, 3, 3, 1, 1, 1, stream);
    launch_conv(tmpBig, ch_b[2 * i + 1], c0cat,  c0cat,  32, 256, 256, 32, 256, 256, 3, 3, 1, 1, 1, stream);
  }

  // ---- c0 -> polar (32,1024) ----
  int nv = 32 * HWp0;
  vox_init_kernel<<<blk(nv), TPB, 0, stream>>>(vox, nv);
  sample_voxmax_kernel<<<blk(N), TPB, 0, stream>>>(c0cat, 32, 256, 256, ccoord, 0.5f,
                                                   pcoord, 0.5f, vox, 32, 1024, N);
  vox_finalize_kernel<<<blk(nv), TPB, 0, stream>>>(vox, c0p, nv);

  // ---- polar_header: 32->32, stride 1, + 2 blocks ----
  launch_conv(c0p, ph_down, nullptr, p0, 32, 32, 1024, 32, 32, 1024, 3, 3, 1, 1, 1, stream);
  for (int i = 0; i < 2; ++i) {
    launch_conv(p0, ph_b[2 * i],     nullptr, tmpBig, 32, 32, 1024, 32, 32, 1024, 3, 3, 1, 1, 1, stream);
    launch_conv(tmpBig, ph_b[2 * i + 1], p0,  p0,     32, 32, 1024, 32, 32, 1024, 3, 3, 1, 1, 1, stream);
  }

  // ---- p0 -> cart (256,256), into channels 32..63 of c0cat ----
  nv = 32 * HW256;
  vox_init_kernel<<<blk(nv), TPB, 0, stream>>>(vox, nv);
  sample_voxmax_kernel<<<blk(N), TPB, 0, stream>>>(p0, 32, 32, 1024, pcoord, 0.5f,
                                                   ccoord, 0.5f, vox, 256, 256, N);
  vox_finalize_kernel<<<blk(nv), TPB, 0, stream>>>(vox, c0cat + (size_t)32 * HW256, nv);

  // ---- cart_res1: 64->64, stride 2, 256^2 -> 128^2, + 4 blocks ----
  launch_conv(c0cat, cr1_down, nullptr, c1cat, 64, 256, 256, 64, 128, 128, 3, 3, 2, 1, 1, stream);
  for (int i = 0; i < 4; ++i) {
    launch_conv(c1cat, cr1_b[2 * i],     nullptr, tmpBig, 64, 128, 128, 64, 128, 128, 3, 3, 1, 1, 1, stream);
    launch_conv(tmpBig, cr1_b[2 * i + 1], c1cat,  c1cat,  64, 128, 128, 64, 128, 128, 3, 3, 1, 1, 1, stream);
  }

  // ---- c1 -> polar (16,512) ----
  nv = 64 * HWp1;
  vox_init_kernel<<<blk(nv), TPB, 0, stream>>>(vox, nv);
  sample_voxmax_kernel<<<blk(N), TPB, 0, stream>>>(c1cat, 64, 128, 128, ccoord, 0.25f,
                                                   pcoord, 0.25f, vox, 16, 512, N);
  vox_finalize_kernel<<<blk(nv), TPB, 0, stream>>>(vox, c1p, nv);

  // ---- polar_res1: 64->64, stride 1, + 3 blocks ----
  launch_conv(c1p, pr1_down, nullptr, p1, 64, 16, 512, 64, 16, 512, 3, 3, 1, 1, 1, stream);
  for (int i = 0; i < 3; ++i) {
    launch_conv(p1, pr1_b[2 * i],     nullptr, tmpBig, 64, 16, 512, 64, 16, 512, 3, 3, 1, 1, 1, stream);
    launch_conv(tmpBig, pr1_b[2 * i + 1], p1,  p1,     64, 16, 512, 64, 16, 512, 3, 3, 1, 1, 1, stream);
  }

  // ---- p1 -> cart (128,128), into channels 64..127 of c1cat ----
  nv = 64 * HW128;
  vox_init_kernel<<<blk(nv), TPB, 0, stream>>>(vox, nv);
  sample_voxmax_kernel<<<blk(N), TPB, 0, stream>>>(p1, 64, 16, 512, pcoord, 0.25f,
                                                   ccoord, 0.25f, vox, 128, 128, N);
  vox_finalize_kernel<<<blk(nv), TPB, 0, stream>>>(vox, c1cat + (size_t)64 * HW128, nv);

  // ---- cart_res2: 128->128, stride 2, 128^2 -> 64^2, + 5 blocks ----
  launch_conv(c1cat, cr2_down, nullptr, c2a, 128, 128, 128, 128, 64, 64, 3, 3, 2, 1, 1, stream);
  for (int i = 0; i < 5; ++i) {
    launch_conv(c2a, cr2_b[2 * i],     nullptr, tmpBig, 128, 64, 64, 128, 64, 64, 3, 3, 1, 1, 1, stream);
    launch_conv(tmpBig, cr2_b[2 * i + 1], c2a,  c2a,    128, 64, 64, 128, 64, 64, 3, 3, 1, 1, 1, stream);
  }

  // ---- add_fuse: relu(conv1x1(c2 + deep_64)) ----
  add_inplace_kernel<<<blk(128 * HW64), TPB, 0, stream>>>(c2a, deep, 128 * HW64);
  launch_conv(c2a, w_fuse, nullptr, c2b, 128, 64, 64, 128, 64, 64, 1, 1, 1, 0, 1, stream);

  // ---- build concat(res0, res1, res2) at 256x256 ----
  hipMemcpyAsync(cat320, c0cat, (size_t)64 * HW256 * sizeof(float),
                 hipMemcpyDeviceToDevice, stream);
  resize_kernel<<<blk(128 * HW256), TPB, 0, stream>>>(c1cat, 128, 128, 128,
                                                      cat320 + (size_t)64 * HW256, 256, 256);
  resize_kernel<<<blk(128 * HW256), TPB, 0, stream>>>(c2b, 128, 64, 64,
                                                      cat320 + (size_t)192 * HW256, 256, 256);

  // ---- output offsets ----
  float* out   = (float*)d_out;
  float* o_oap = out;                              // (64,N)
  float* o_prp = out + (size_t)64 * N;             // (64,N)
  float* o_r0  = out + (size_t)128 * N;            // (3,256,256)
  float* o_r1  = o_r0 + (size_t)3 * HW256;
  float* o_r2  = o_r1 + (size_t)3 * HW256;
  float* o_c2  = o_r2 + (size_t)3 * HW256;         // (128,64,64)

  // aux1 consumes c0cat, then c0cat is recycled for out_conv2's result
  aux_head_kernel<<<blk(HW256), TPB, 0, stream>>>(c0cat, 64, HW256, aux1w, aux1b, o_r0);

  // ---- out convs: 320->128->64 at 256^2 ----
  launch_conv(cat320, w_oc1, nullptr, tmpBig, 320, 256, 256, 128, 256, 256, 3, 3, 1, 1, 1, stream);
  launch_conv(tmpBig, w_oc2, nullptr, c0cat, 128, 256, 256, 64, 256, 256, 3, 3, 1, 1, 1, stream);

  // ---- point outputs ----
  sample_points_kernel<<<blk(N), TPB, 0, stream>>>(c0cat, 64, 256, 256, ccoord, 0.5f, o_oap, N);
  sample_points_kernel<<<blk(N), TPB, 0, stream>>>(p1, 64, 16, 512, pcoord, 0.5f, o_prp, N);

  // ---- aux heads 2/3 + final c2 ----
  aux_head_kernel<<<blk(HW256), TPB, 0, stream>>>(cat320 + (size_t)64 * HW256, 128, HW256,
                                                  aux2w, aux2b, o_r1);
  aux_head_kernel<<<blk(HW256), TPB, 0, stream>>>(cat320 + (size_t)192 * HW256, 128, HW256,
                                                  aux3w, aux3b, o_r2);
  hipMemcpyAsync(o_c2, c2b, (size_t)128 * HW64 * sizeof(float),
                 hipMemcpyDeviceToDevice, stream);
}
